// VAEAKFCombinedLinear_4123168604714
// MI455X (gfx1250) — matrix operand order repaired and checked by reading, not compiled, not run on hardware
//
#include <hip/hip_runtime.h>
#include <hip/hip_bf16.h>
#include <math.h>

typedef __attribute__((ext_vector_type(16))) _Float16 v16h;
typedef __attribute__((ext_vector_type(8)))  _Float16 v8h;
typedef __attribute__((ext_vector_type(8)))  float    v8f;

constexpr int kL   = 256;   // sequence length
constexpr int kB   = 1024;  // batch
constexpr int kInp = 32;
constexpr int kObs = 64;
constexpr int kSt  = 128;
constexpr int kK   = 256;
constexpr int kNL  = 8;

__device__ __forceinline__ v8f zero8() {
  v8f z;
#pragma unroll
  for (int i = 0; i < 8; ++i) z[i] = 0.f;
  return z;
}

__device__ __forceinline__ v8f wmma16(v16h a, v16h b, v8f c) {
  return __builtin_amdgcn_wmma_f32_16x16x32_f16(false, a, false, b, (short)0, c,
                                                false, false);
}

__device__ __forceinline__ float sigf(float x) { return 1.f / (1.f + expf(-x)); }

// ---- A fragment: 16x32 f16 tile, row-major source with row stride lda (halves).
// Lane l: row m = l%16, kh = l/16. K = 16*kh+e.
__device__ __forceinline__ v16h load_frag_a(const _Float16* base, int lda, int lane) {
  int m = lane & 15, kh = lane >> 4;
  const _Float16* p = base + (size_t)m * lda + 16 * kh;
  v8h lo = *(const v8h*)(p);
  v8h hi = *(const v8h*)(p + 8);
  v16h a;
#pragma unroll
  for (int e = 0; e < 8; ++e) { a[e] = lo[e]; a[8 + e] = hi[e]; }
  return a;
}

// ---- B fragment from pre-packed weights: per (nTile,kTile) fragment, lane-major,
// 16 contiguous f16 per lane (K = kTile*32 + 16*(lane/16) + e, N = nTile*16 + lane%16).
__device__ __forceinline__ v16h load_frag_b(const _Float16* P, int kTiles,
                                            int nTile, int kTile, int lane) {
  const _Float16* p = P + ((((size_t)nTile * kTiles + kTile) * 32 + lane) << 4);
  v8h lo = *(const v8h*)(p);
  v8h hi = *(const v8h*)(p + 8);
  v16h b;
#pragma unroll
  for (int e = 0; e < 8; ++e) { b[e] = lo[e]; b[8 + e] = hi[e]; }
  return b;
}

// ================= weight packing: f32 [N,K] row-major -> f16 B-fragment order ====
__global__ void pack_w(const float* __restrict__ W, _Float16* __restrict__ P,
                       int N, int K, int kTiles, int nTiles) {
  int idx = blockIdx.x * 256 + threadIdx.x;
  int total = nTiles * kTiles * 512;
  if (idx >= total) return;
  int e = idx & 15;
  int lane = (idx >> 4) & 31;
  int t = idx >> 9;
  int kT = t % kTiles;
  int nT = t / kTiles;
  int k = kT * 32 + 16 * (lane >> 4) + e;
  int n = nT * 16 + (lane & 15);
  float v = (k < K && n < N) ? W[(size_t)n * K + k] : 0.f;
  P[idx] = (_Float16)v;
}

// ================= initial gauss-decoder head (batch-invariant) ==================
__global__ void init_state_head(const float* __restrict__ h0,
                                const float* __restrict__ W1, const float* __restrict__ b1,
                                const float* __restrict__ W2, const float* __restrict__ b2,
                                const float* __restrict__ Wmu, const float* __restrict__ bmu,
                                const float* __restrict__ Wls, const float* __restrict__ bls,
                                float* __restrict__ mu0, float* __restrict__ ls0) {
  __shared__ float tv[2 * kK];
  int j = threadIdx.x;  // 512 threads
  float a = 0.f, bb = 0.f;
  for (int k = 0; k < kK; ++k) {
    float h = h0[k];
    a += h * W1[(size_t)j * kK + k];
    bb += h * W2[(size_t)j * kK + k];
  }
  tv[j] = tanhf(a + b1[j]) * sigf(bb + b2[j]);
  __syncthreads();
  if (j < kSt) {
    float m = bmu[j], l = bls[j];
    for (int k = 0; k < 2 * kK; ++k) {
      float t = tv[k];
      m += t * Wmu[(size_t)j * 2 * kK + k];
      l += t * Wls[(size_t)j * 2 * kK + k];
    }
    mu0[j] = m;
    ls0[j] = l;
  }
}

__global__ void make_state0(const float* __restrict__ mu0, const float* __restrict__ ls0,
                            const float* __restrict__ eps, _Float16* __restrict__ s0) {
  int i = blockIdx.x * 256 + threadIdx.x;  // kB*kSt
  if (i >= kB * kSt) return;
  int s = i & (kSt - 1);
  s0[i] = (_Float16)(mu0[s] + expf(0.5f * ls0[s]) * eps[i]);
}

// ================= prior scan (weight-LSTM + mixture linears), 64 rows / WG ======
// 512 threads = 16 waves. Gate tiles: 64 positions -> 4/wave (cfr[4] persistent).
// h tile double-buffered in LDS; state & u double-buffered; c lives in VGPRs.
__global__ __launch_bounds__(512) void prior_scan(
    const _Float16* __restrict__ s0,       // [kB,128]
    const float* __restrict__ u_seq,       // [kL,kB,32]
    const float* __restrict__ eps_seq,     // [kL,kB,128]
    const _Float16* __restrict__ Pih,      // packed K=128,N=1024 (kT=4)
    const _Float16* __restrict__ Phh,      // packed K=256,N=1024 (kT=8)
    const _Float16* __restrict__ PwlW,     // packed K=256,N=16  (kT=8)
    const _Float16* __restrict__ Plin,     // packed K=160,N=1024 (kT=5)
    const float* __restrict__ bih, const float* __restrict__ bhh,
    const float* __restrict__ wl_b, const float* __restrict__ lin_b,
    const float* __restrict__ dyn_ls, const float* __restrict__ dyn_h0,
    _Float16* __restrict__ sampled,        // [kL,kB,128] f16
    float* __restrict__ wm_out,            // [kL,kB,8]
    float* __restrict__ wh_out, float* __restrict__ wc_out) {  // [kB,256]
  __shared__ __align__(16) _Float16 st[2][64 * kSt];
  __shared__ __align__(16) _Float16 ub[2][64 * kInp];
  __shared__ __align__(16) _Float16 whb[2][64 * kK];
  __shared__ float wlog[64 * 16];
  __shared__ float wsm[64 * kNL];

  const int tid = threadIdx.x;
  const int lane = tid & 31, wave = tid >> 5;
  const int l15 = lane & 15;
  const int mbase = blockIdx.x * 64;

  for (int i = tid; i < 64 * kSt; i += 512)
    st[0][i] = s0[(size_t)(mbase + (i >> 7)) * kSt + (i & 127)];
  for (int i = tid; i < 64 * kInp; i += 512)
    ub[0][i] = (_Float16)u_seq[(size_t)(mbase + (i >> 5)) * kInp + (i & 31)];
  for (int i = tid; i < 64 * kK; i += 512)
    whb[0][i] = (_Float16)dyn_h0[i & 255];

  // ---- hoist t-invariant scalars ----
  float gb[4][4];      // gate biases per position
#pragma unroll
  for (int p = 0; p < 4; ++p) {
    int nt = (wave * 4 + p) & 15;
    int n = nt * 16 + l15;
#pragma unroll
    for (int g = 0; g < 4; ++g) gb[p][g] = bih[g * kK + n] + bhh[g * kK + n];
  }
  float dsc[2], lb[2][kNL];
#pragma unroll
  for (int p = 0; p < 2; ++p) {
    int stile = (wave * 2 + p) & 7;
    int s = stile * 16 + l15;
    dsc[p] = expf(0.5f * dyn_ls[s]);
#pragma unroll
    for (int n = 0; n < kNL; ++n) lb[p][n] = lin_b[n * kSt + s];
  }
  float wlb = (l15 < kNL) ? wl_b[l15] : 0.f;

  v8f cfr[4];
#pragma unroll
  for (int p = 0; p < 4; ++p) cfr[p] = zero8();

  for (int t = 0; t < kL; ++t) {
    const int cur = t & 1, nxt = cur ^ 1;
    __syncthreads();
    // ---- gates: z = state @ Wih^T + h @ Whh^T ; write new h to whb[nxt] ----
#pragma unroll 1
    for (int p = 0; p < 4; ++p) {
      int pos = wave * 4 + p;
      int mt = pos >> 4, nt = pos & 15;
      v8f ai = zero8(), af = zero8(), ag = zero8(), ao = zero8();
#pragma unroll
      for (int kc = 0; kc < 4; ++kc) {
        v16h a = load_frag_a(&st[cur][mt * 16 * kSt + kc * 32], kSt, lane);
        ai = wmma16(a, load_frag_b(Pih, 4, 0 * 16 + nt, kc, lane), ai);
        af = wmma16(a, load_frag_b(Pih, 4, 1 * 16 + nt, kc, lane), af);
        ag = wmma16(a, load_frag_b(Pih, 4, 2 * 16 + nt, kc, lane), ag);
        ao = wmma16(a, load_frag_b(Pih, 4, 3 * 16 + nt, kc, lane), ao);
      }
#pragma unroll
      for (int kc = 0; kc < 8; ++kc) {
        v16h a = load_frag_a(&whb[cur][mt * 16 * kK + kc * 32], kK, lane);
        ai = wmma16(a, load_frag_b(Phh, 8, 0 * 16 + nt, kc, lane), ai);
        af = wmma16(a, load_frag_b(Phh, 8, 1 * 16 + nt, kc, lane), af);
        ag = wmma16(a, load_frag_b(Phh, 8, 2 * 16 + nt, kc, lane), ag);
        ao = wmma16(a, load_frag_b(Phh, 8, 3 * 16 + nt, kc, lane), ao);
      }
      int n = nt * 16 + l15;
#pragma unroll
      for (int r = 0; r < 8; ++r) {
        float ig = sigf(ai[r] + gb[p][0]), fg = sigf(af[r] + gb[p][1]);
        float gg = tanhf(ag[r] + gb[p][2]), og = sigf(ao[r] + gb[p][3]);
        float c = fg * cfr[p][r] + ig * gg;
        cfr[p][r] = c;
        float h = og * tanhf(c);
        int m = mt * 16 + r + 8 * (lane >> 4);
        whb[nxt][m * kK + n] = (_Float16)h;
        if (t == kL - 1) {
          wh_out[(size_t)(mbase + m) * kK + n] = h;
          wc_out[(size_t)(mbase + m) * kK + n] = cfr[p][r];
        }
      }
    }
    __syncthreads();
    // ---- mixture logits: new_h @ wl_W^T (N padded to 16) ----
    if (wave < 4) {
      int mt = wave;
      v8f acc = zero8();
#pragma unroll
      for (int kc = 0; kc < 8; ++kc) {
        v16h a = load_frag_a(&whb[nxt][mt * 16 * kK + kc * 32], kK, lane);
        acc = wmma16(a, load_frag_b(PwlW, 8, 0, kc, lane), acc);
      }
#pragma unroll
      for (int r = 0; r < 8; ++r) {
        int m = mt * 16 + r + 8 * (lane >> 4);
        wlog[m * 16 + l15] = acc[r] + wlb;
      }
    }
    // stage u(t+1)
    if (t + 1 < kL)
      for (int i = tid; i < 64 * kInp; i += 512)
        ub[nxt][i] =
            (_Float16)u_seq[((size_t)(t + 1) * kB + mbase + (i >> 5)) * kInp + (i & 31)];
    __syncthreads();
    // ---- softmax over 8 mixture weights per row ----
    if (tid < 64) {
      float l[kNL], mx = -1e30f;
      for (int n = 0; n < kNL; ++n) { l[n] = wlog[tid * 16 + n]; mx = fmaxf(mx, l[n]); }
      float s = 0.f;
      for (int n = 0; n < kNL; ++n) { l[n] = expf(l[n] - mx); s += l[n]; }
      float inv = 1.f / s;
      for (int n = 0; n < kNL; ++n) {
        float w = l[n] * inv;
        wsm[tid * kNL + n] = w;
        wm_out[((size_t)t * kB + mbase + tid) * kNL + n] = w;
      }
    }
    __syncthreads();
    // ---- mus = [state|u] @ lin_W^T ; mu = sum_n w_n * mus_n ; sample state ----
#pragma unroll 1
    for (int p = 0; p < 2; ++p) {
      int pos = wave * 2 + p;        // 32 positions
      int mt = pos >> 3, stile = pos & 7;
      v8f mu8 = zero8();
#pragma unroll 1
      for (int g = 0; g < 2; ++g) {  // mixture components in 2 groups of 4
        v8f acc[4];
#pragma unroll
        for (int j = 0; j < 4; ++j) acc[j] = zero8();
#pragma unroll
        for (int kc = 0; kc < 5; ++kc) {
          v16h a = (kc < 4)
                       ? load_frag_a(&st[cur][mt * 16 * kSt + kc * 32], kSt, lane)
                       : load_frag_a(&ub[cur][mt * 16 * kInp], kInp, lane);
#pragma unroll
          for (int j = 0; j < 4; ++j)
            acc[j] = wmma16(a, load_frag_b(Plin, 5, (g * 4 + j) * 8 + stile, kc, lane),
                            acc[j]);
        }
#pragma unroll
        for (int j = 0; j < 4; ++j) {
          int n = g * 4 + j;
#pragma unroll
          for (int r = 0; r < 8; ++r) {
            int m = mt * 16 + r + 8 * (lane >> 4);
            mu8[r] += wsm[m * kNL + n] * (acc[j][r] + lb[p][n]);
          }
        }
      }
      int s = stile * 16 + l15;
#pragma unroll
      for (int r = 0; r < 8; ++r) {
        int m = mt * 16 + r + 8 * (lane >> 4);
        float sv = mu8[r] + dsc[p] * eps_seq[((size_t)t * kB + mbase + m) * kSt + s];
        _Float16 sh = (_Float16)sv;
        st[nxt][m * kSt + s] = sh;
        sampled[((size_t)t * kB + mbase + m) * kSt + s] = sh;
      }
    }
  }
}

// ================= emission: obs = state @ H^T + b ; sample ; build allx =========
__global__ __launch_bounds__(256) void emission(
    const _Float16* __restrict__ sampled,  // [kL*kB,128]
    const float* __restrict__ u_seq,       // [kL*kB,32]
    const float* __restrict__ eps_obs,     // [kL*kB,64]
    const _Float16* __restrict__ PH,       // packed K=128,N=64 (kT=4)
    const float* __restrict__ ebias, const float* __restrict__ logdelta,
    float* __restrict__ obs_mu, float* __restrict__ obs_sample,
    _Float16* __restrict__ allx) {         // [kL*kB,96]
  int tid = threadIdx.x, lane = tid & 31, wave = tid >> 5;
  size_t r0 = (size_t)blockIdx.x * 64;
  for (int i = tid; i < 64 * kInp; i += 256)
    allx[(r0 + (i >> 5)) * 96 + (i & 31)] =
        (_Float16)u_seq[(r0 + (i >> 5)) * kInp + (i & 31)];
#pragma unroll 1
  for (int p = 0; p < 2; ++p) {
    int pos = wave * 2 + p;
    int mt = pos >> 2, nt = pos & 3;
    v8f acc = zero8();
#pragma unroll
    for (int kc = 0; kc < 4; ++kc) {
      v16h a = load_frag_a(sampled + (r0 + mt * 16) * kSt + kc * 32, kSt, lane);
      acc = wmma16(a, load_frag_b(PH, 4, nt, kc, lane), acc);
    }
    int o = nt * 16 + (lane & 15);
    float bb = ebias[o], dl = expf(0.5f * logdelta[o]);
#pragma unroll
    for (int r = 0; r < 8; ++r) {
      size_t row = r0 + mt * 16 + r + 8 * (lane >> 4);
      float mu = acc[r] + bb;
      float sm = mu + dl * eps_obs[row * kObs + o];
      obs_mu[row * kObs + o] = mu;
      obs_sample[row * kObs + o] = sm;
      allx[row * 96 + 32 + o] = (_Float16)sm;
    }
  }
}

// ================= preprocess: px = relu(relu(allx W1^T) W2^T) ===================
__global__ __launch_bounds__(256) void preprocess(
    const _Float16* __restrict__ allx,  // [kL*kB,96]
    const _Float16* __restrict__ P1,    // packed K=96,N=256 (kT=3)
    const _Float16* __restrict__ P2,    // packed K=256,N=256 (kT=8)
    const float* __restrict__ b1, const float* __restrict__ b2,
    _Float16* __restrict__ px) {        // [kL*kB,256]
  __shared__ __align__(16) _Float16 h1[64 * kK];
  int tid = threadIdx.x, lane = tid & 31, wave = tid >> 5;
  size_t r0 = (size_t)blockIdx.x * 64;
#pragma unroll 1
  for (int p = 0; p < 8; ++p) {
    int pos = wave * 8 + p;
    int mt = pos >> 4, nt = pos & 15;
    v8f acc = zero8();
#pragma unroll
    for (int kc = 0; kc < 3; ++kc) {
      v16h a = load_frag_a(allx + (r0 + mt * 16) * 96 + kc * 32, 96, lane);
      acc = wmma16(a, load_frag_b(P1, 3, nt, kc, lane), acc);
    }
    int n = nt * 16 + (lane & 15);
    float bb = b1[n];
#pragma unroll
    for (int r = 0; r < 8; ++r) {
      int m = mt * 16 + r + 8 * (lane >> 4);
      h1[m * kK + n] = (_Float16)fmaxf(acc[r] + bb, 0.f);
    }
  }
  __syncthreads();
#pragma unroll 1
  for (int p = 0; p < 8; ++p) {
    int pos = wave * 8 + p;
    int mt = pos >> 4, nt = pos & 15;
    v8f acc = zero8();
#pragma unroll
    for (int kc = 0; kc < 8; ++kc) {
      v16h a = load_frag_a(&h1[mt * 16 * kK + kc * 32], kK, lane);
      acc = wmma16(a, load_frag_b(P2, 8, nt, kc, lane), acc);
    }
    int n = nt * 16 + (lane & 15);
    float bb = b2[n];
#pragma unroll
    for (int r = 0; r < 8; ++r) {
      size_t m = r0 + mt * 16 + r + 8 * (lane >> 4);
      px[m * kK + n] = (_Float16)fmaxf(acc[r] + bb, 0.f);
    }
  }
}

// ================= posterior LSTM scan (512 threads, h double-buffered) =========
__global__ __launch_bounds__(512) void posterior_scan(
    const _Float16* __restrict__ px,   // [kL,kB,256]
    const float* __restrict__ h0vec,   // [256]
    const _Float16* __restrict__ Pih,  // packed K=256,N=1024 (kT=8)
    const _Float16* __restrict__ Phh,  // packed K=256,N=1024 (kT=8)
    const float* __restrict__ bih, const float* __restrict__ bhh,
    float* __restrict__ h_out, float* __restrict__ c_out) {  // [kB,256]
  __shared__ __align__(16) _Float16 hb[2][64 * kK];
  int tid = threadIdx.x, lane = tid & 31, wave = tid >> 5;
  const int l15 = lane & 15;
  int mbase = blockIdx.x * 64;
  for (int i = tid; i < 64 * kK; i += 512) hb[0][i] = (_Float16)h0vec[i & 255];

  float gb[4][4];
#pragma unroll
  for (int p = 0; p < 4; ++p) {
    int nt = (wave * 4 + p) & 15;
    int n = nt * 16 + l15;
#pragma unroll
    for (int g = 0; g < 4; ++g) gb[p][g] = bih[g * kK + n] + bhh[g * kK + n];
  }

  v8f cfr[4];
#pragma unroll
  for (int p = 0; p < 4; ++p) cfr[p] = zero8();

  for (int t = 0; t < kL; ++t) {
    const int cur = t & 1, nxt = cur ^ 1;
    __syncthreads();
#pragma unroll 1
    for (int p = 0; p < 4; ++p) {
      int pos = wave * 4 + p;
      int mt = pos >> 4, nt = pos & 15;
      v8f ai = zero8(), af = zero8(), ag = zero8(), ao = zero8();
#pragma unroll
      for (int kc = 0; kc < 8; ++kc) {
        v16h a = load_frag_a(
            px + ((size_t)t * kB + mbase + mt * 16) * kK + kc * 32, kK, lane);
        ai = wmma16(a, load_frag_b(Pih, 8, 0 * 16 + nt, kc, lane), ai);
        af = wmma16(a, load_frag_b(Pih, 8, 1 * 16 + nt, kc, lane), af);
        ag = wmma16(a, load_frag_b(Pih, 8, 2 * 16 + nt, kc, lane), ag);
        ao = wmma16(a, load_frag_b(Pih, 8, 3 * 16 + nt, kc, lane), ao);
      }
#pragma unroll
      for (int kc = 0; kc < 8; ++kc) {
        v16h a = load_frag_a(&hb[cur][mt * 16 * kK + kc * 32], kK, lane);
        ai = wmma16(a, load_frag_b(Phh, 8, 0 * 16 + nt, kc, lane), ai);
        af = wmma16(a, load_frag_b(Phh, 8, 1 * 16 + nt, kc, lane), af);
        ag = wmma16(a, load_frag_b(Phh, 8, 2 * 16 + nt, kc, lane), ag);
        ao = wmma16(a, load_frag_b(Phh, 8, 3 * 16 + nt, kc, lane), ao);
      }
      int n = nt * 16 + l15;
#pragma unroll
      for (int r = 0; r < 8; ++r) {
        float ig = sigf(ai[r] + gb[p][0]), fg = sigf(af[r] + gb[p][1]);
        float gg = tanhf(ag[r] + gb[p][2]), og = sigf(ao[r] + gb[p][3]);
        float c = fg * cfr[p][r] + ig * gg;
        cfr[p][r] = c;
        float h = og * tanhf(c);
        int m = mt * 16 + r + 8 * (lane >> 4);
        hb[nxt][m * kK + n] = (_Float16)h;
        if (t == kL - 1) {
          h_out[(size_t)(mbase + m) * kK + n] = h;
          c_out[(size_t)(mbase + m) * kK + n] = cfr[p][r];
        }
      }
    }
  }
}

// ==================================== launch ====================================
extern "C" void kernel_launch(void* const* d_in, const int* in_sizes, int n_in,
                              void* d_out, int out_size, void* d_ws, size_t ws_size,
                              hipStream_t stream) {
  (void)in_sizes; (void)n_in; (void)out_size; (void)ws_size;
  const float* u_seq      = (const float*)d_in[0];
  const float* eps_init   = (const float*)d_in[1];
  const float* eps_seq    = (const float*)d_in[2];
  const float* eps_obs    = (const float*)d_in[3];
  const float* pp_W1      = (const float*)d_in[4];
  const float* pp_b1      = (const float*)d_in[5];
  const float* pp_W2      = (const float*)d_in[6];
  const float* pp_b2      = (const float*)d_in[7];
  const float* rnn_Wih    = (const float*)d_in[8];
  const float* rnn_Whh    = (const float*)d_in[9];
  const float* rnn_bih    = (const float*)d_in[10];
  const float* rnn_bhh    = (const float*)d_in[11];
  const float* post_h0    = (const float*)d_in[12];
  const float* gd_W1      = (const float*)d_in[13];
  const float* gd_b1      = (const float*)d_in[14];
  const float* gd_W2      = (const float*)d_in[15];
  const float* gd_b2      = (const float*)d_in[16];
  const float* gd_Wmu     = (const float*)d_in[17];
  const float* gd_bmu     = (const float*)d_in[18];
  const float* gd_Wls     = (const float*)d_in[19];
  const float* gd_bls     = (const float*)d_in[20];
  const float* dyn_h0     = (const float*)d_in[21];
  const float* wl_Wih     = (const float*)d_in[22];
  const float* wl_Whh     = (const float*)d_in[23];
  const float* wl_bih     = (const float*)d_in[24];
  const float* wl_bhh     = (const float*)d_in[25];
  const float* wl_W       = (const float*)d_in[26];
  const float* wl_b       = (const float*)d_in[27];
  const float* lin_W      = (const float*)d_in[28];
  const float* lin_b      = (const float*)d_in[29];
  const float* dyn_ls     = (const float*)d_in[30];
  const float* e_logdelta = (const float*)d_in[31];
  const float* e_H        = (const float*)d_in[32];
  const float* e_bias     = (const float*)d_in[33];

  // ---- output layout (return order, flat floats) ----
  float* out = (float*)d_out;
  float* o_obs_mu     = out;
  float* o_obs_sample = o_obs_mu + (size_t)kL * kB * kObs;
  float* o_hpost      = o_obs_sample + (size_t)kL * kB * kObs;
  float* o_cpost      = o_hpost + (size_t)kB * kK;
  float* o_wh         = o_cpost + (size_t)kB * kK;
  float* o_wc         = o_wh + (size_t)kB * kK;
  float* o_wm         = o_wc + (size_t)kB * kK;

  // ---- workspace carve-up ----
  char* ws = (char*)d_ws;
  size_t off = 0;
  auto take = [&](size_t bytes) -> void* {
    void* p = ws + off;
    off = (off + bytes + 255) & ~(size_t)255;
    return p;
  };
  _Float16* P_wih = (_Float16*)take((size_t)128 * 1024 * 2);
  _Float16* P_whh = (_Float16*)take((size_t)256 * 1024 * 2);
  _Float16* P_wlW = (_Float16*)take((size_t)256 * 16 * 2);
  _Float16* P_lin = (_Float16*)take((size_t)160 * 1024 * 2);
  _Float16* P_rih = (_Float16*)take((size_t)256 * 1024 * 2);
  _Float16* P_rhh = (_Float16*)take((size_t)256 * 1024 * 2);
  _Float16* P_pp1 = (_Float16*)take((size_t)96 * 256 * 2);
  _Float16* P_pp2 = (_Float16*)take((size_t)256 * 256 * 2);
  _Float16* P_H   = (_Float16*)take((size_t)128 * 64 * 2);
  float*    mu0   = (float*)take(kSt * 4);
  float*    ls0   = (float*)take(kSt * 4);
  _Float16* s0    = (_Float16*)take((size_t)kB * kSt * 2);
  _Float16* smp   = (_Float16*)take((size_t)kL * kB * kSt * 2);
  _Float16* allx  = (_Float16*)take((size_t)kL * kB * 96 * 2);
  _Float16* px    = (_Float16*)take((size_t)kL * kB * kK * 2);

  auto packGrid = [](int nT, int kT) { return (nT * kT * 512 + 255) / 256; };
  pack_w<<<packGrid(64, 4), 256, 0, stream>>>(wl_Wih, P_wih, 1024, 128, 4, 64);
  pack_w<<<packGrid(64, 8), 256, 0, stream>>>(wl_Whh, P_whh, 1024, 256, 8, 64);
  pack_w<<<packGrid(1, 8),  256, 0, stream>>>(wl_W,   P_wlW, 8,    256, 8, 1);
  pack_w<<<packGrid(64, 5), 256, 0, stream>>>(lin_W,  P_lin, 1024, 160, 5, 64);
  pack_w<<<packGrid(64, 8), 256, 0, stream>>>(rnn_Wih, P_rih, 1024, 256, 8, 64);
  pack_w<<<packGrid(64, 8), 256, 0, stream>>>(rnn_Whh, P_rhh, 1024, 256, 8, 64);
  pack_w<<<packGrid(16, 3), 256, 0, stream>>>(pp_W1,  P_pp1, 256, 96, 3, 16);
  pack_w<<<packGrid(16, 8), 256, 0, stream>>>(pp_W2,  P_pp2, 256, 256, 8, 16);
  pack_w<<<packGrid(4, 4),  256, 0, stream>>>(e_H,    P_H,   64,  128, 4, 4);

  init_state_head<<<1, 512, 0, stream>>>(post_h0, gd_W1, gd_b1, gd_W2, gd_b2,
                                         gd_Wmu, gd_bmu, gd_Wls, gd_bls, mu0, ls0);
  make_state0<<<(kB * kSt + 255) / 256, 256, 0, stream>>>(mu0, ls0, eps_init, s0);

  prior_scan<<<kB / 64, 512, 0, stream>>>(
      s0, u_seq, eps_seq, P_wih, P_whh, P_wlW, P_lin, wl_bih, wl_bhh, wl_b,
      lin_b, dyn_ls, dyn_h0, smp, o_wm, o_wh, o_wc);

  emission<<<(kL * kB) / 64, 256, 0, stream>>>(smp, u_seq, eps_obs, P_H, e_bias,
                                               e_logdelta, o_obs_mu, o_obs_sample,
                                               allx);

  preprocess<<<(kL * kB) / 64, 256, 0, stream>>>(allx, P_pp1, P_pp2, pp_b1,
                                                 pp_b2, px);

  posterior_scan<<<kB / 64, 512, 0, stream>>>(px, post_h0, P_rih, P_rhh, rnn_bih,
                                              rnn_bhh, o_hpost, o_cpost);
}
